// MultiLayerLSTM_40604620816886
// MI455X (gfx1250) — matrix operand order repaired and checked by reading, not compiled, not run on hardware
//
#include <hip/hip_runtime.h>
#include <hip/hip_bf16.h>

// ---------------------------------------------------------------------------
// 2-layer LSTM, T=512, B=64, IN=H=1024, fp32 in/out.
// Persistent-kernel design: 128 blocks (64 per layer), software-pipelined
// layers, 1 grid barrier per timestep. Split-bf16 (hi+lo) WMMA GEMMs
// (v_wmma_f32_16x16x32_bf16) for ~fp32 accuracy. Cell state c lives in
// registers for the whole sequence; h ping-pongs via small global buffers
// pre-split into bf16 hi/lo. Weights pre-split to bf16 hi/lo in workspace
// (64 MB, L2-resident on MI455X's 192 MB L2).
// ---------------------------------------------------------------------------

#define T_STEPS 512
#define BATCH   64
#define HID     1024
#define NE      (BATCH * HID)        // 65536 elements per [B,H] plane
#define WE      (4 * HID * HID)      // 4194304 elements per weight matrix

typedef __attribute__((ext_vector_type(8)))  __bf16 bf16x8;
typedef __attribute__((ext_vector_type(16))) __bf16 bf16x16;
typedef __attribute__((ext_vector_type(8)))  float  v8f;

// Split fp32 into truncated-bf16 hi + bf16(lo) residual. hi is exact in bf16;
// hi*hi + hi*lo + lo*hi recovers ~fp32 products.
__device__ inline void splitbf(float x, unsigned short& hi, unsigned short& lo) {
    unsigned u = __float_as_uint(x);
    hi = (unsigned short)(u >> 16);
    float r = x - __uint_as_float(u & 0xFFFF0000u);
    lo = (unsigned short)(__float_as_uint(r) >> 16);
}

// A operand (16x32 bf16): lane needs K = base+{0..7} and base+{16..23}
// (lanes 0-15) or shifted by 8 (lanes 16-31). Two 16B loads.
__device__ inline bf16x16 loadA(const unsigned short* p) {
    bf16x8 a = *(const bf16x8*)(p);
    bf16x8 b = *(const bf16x8*)(p + 16);
    return __builtin_shufflevector(a, b, 0,1,2,3,4,5,6,7,8,9,10,11,12,13,14,15);
}
// B operand (32x16 bf16): lane n needs K = base+{0..7} and base+{16..23}
// (lanes 0-15) or shifted by 8 (lanes 16-31). Two 16B loads.
__device__ inline bf16x16 loadB(const unsigned short* p) {
    bf16x8 a = *(const bf16x8*)(p);
    bf16x8 b = *(const bf16x8*)(p + 16);
    return __builtin_shufflevector(a, b, 0,1,2,3,4,5,6,7,8,9,10,11,12,13,14,15);
}

__device__ inline v8f wmma_bf16(bf16x16 a, bf16x16 b, v8f c) {
    // (neg_a, A, neg_b, B, c_mod, C, reuse_a, reuse_b)
    return __builtin_amdgcn_wmma_f32_16x16x32_bf16(false, a, false, b,
                                                   (short)0, c, false, false);
}

// Accumulate C += A*W^T over K=1024 for two gate tiles (split-bf16: 3 WMMA per
// k-tile per output tile).
__device__ inline void gemm_acc(const unsigned short* aHi, const unsigned short* aLo,
                                const unsigned short* wHi, const unsigned short* wLo,
                                int rowA, int wrow0, int wrow1,
                                v8f& acc0, v8f& acc1) {
    const int lane = threadIdx.x & 31;
    const int aoff = (lane & 16) ? 8 : 0;
    const int boff = (lane & 16) ? 8 : 0;
    const unsigned short* pAH  = aHi + rowA  * HID + aoff;
    const unsigned short* pAL  = aLo + rowA  * HID + aoff;
    const unsigned short* pB0H = wHi + wrow0 * HID + boff;
    const unsigned short* pB0L = wLo + wrow0 * HID + boff;
    const unsigned short* pB1H = wHi + wrow1 * HID + boff;
    const unsigned short* pB1L = wLo + wrow1 * HID + boff;
#pragma unroll 4
    for (int kb = 0; kb < HID; kb += 32) {
        bf16x16 aH  = loadA(pAH + kb);
        bf16x16 aL  = loadA(pAL + kb);
        bf16x16 b0H = loadB(pB0H + kb);
        bf16x16 b0L = loadB(pB0L + kb);
        bf16x16 b1H = loadB(pB1H + kb);
        bf16x16 b1L = loadB(pB1L + kb);
        __builtin_prefetch((const void*)(pB0H + kb + 64), 0, 1);
        acc0 = wmma_bf16(aH, b0H, acc0);
        acc0 = wmma_bf16(aH, b0L, acc0);
        acc0 = wmma_bf16(aL, b0H, acc0);
        acc1 = wmma_bf16(aH, b1H, acc1);
        acc1 = wmma_bf16(aH, b1L, acc1);
        acc1 = wmma_bf16(aL, b1H, acc1);
    }
}

// Grid-wide sense-reversal barrier (all 128 blocks resident).
__device__ inline void gridBarrier(unsigned* cnt, unsigned* gen, unsigned nb) {
    __syncthreads();
    if (threadIdx.x == 0) {
        __threadfence();
        unsigned g = __hip_atomic_load(gen, __ATOMIC_RELAXED, __HIP_MEMORY_SCOPE_AGENT);
        unsigned v = __hip_atomic_fetch_add(cnt, 1u, __ATOMIC_ACQ_REL, __HIP_MEMORY_SCOPE_AGENT);
        if (v == nb - 1u) {
            __hip_atomic_store(cnt, 0u, __ATOMIC_RELAXED, __HIP_MEMORY_SCOPE_AGENT);
            __hip_atomic_fetch_add(gen, 1u, __ATOMIC_RELEASE, __HIP_MEMORY_SCOPE_AGENT);
        } else {
            while (__hip_atomic_load(gen, __ATOMIC_ACQUIRE, __HIP_MEMORY_SCOPE_AGENT) == g) {
                __builtin_amdgcn_s_sleep(2);
            }
        }
        __threadfence();
    }
    __syncthreads();
}

// One LSTM step for this block's 16 hidden columns.
// Gates = A@Wih^T + Hprev@Whh^T + b ; c,h update ; h written as bf16 hi/lo
// (and optionally fp32 to out).
__device__ inline void lstm_step(const unsigned short* aHi, const unsigned short* aLo,
                                 const unsigned short* hpHi, const unsigned short* hpLo,
                                 const unsigned short* wihHi, const unsigned short* wihLo,
                                 const unsigned short* whhHi, const unsigned short* whhLo,
                                 const float* bias, int j0,
                                 unsigned short* houtHi, unsigned short* houtLo,
                                 float* foutOpt, float* gate_lds, float c_reg[4]) {
    const int tid  = threadIdx.x;
    const int wid  = tid >> 5;
    const int lane = tid & 31;
    const int m_tile = wid & 3;          // 4 M tiles (batch 64)
    const int gpair  = wid >> 2;         // waves 0-3: gates {i,f}; 4-7: {g,o}
    const int g0 = gpair * 2, g1 = g0 + 1;
    const int rowA  = m_tile * 16 + (lane & 15);
    const int wrow0 = g0 * HID + j0 + (lane & 15);
    const int wrow1 = wrow0 + HID;

    v8f acc0 = {0.f,0.f,0.f,0.f,0.f,0.f,0.f,0.f};
    v8f acc1 = {0.f,0.f,0.f,0.f,0.f,0.f,0.f,0.f};
    gemm_acc(aHi,  aLo,  wihHi, wihLo, rowA, wrow0, wrow1, acc0, acc1);   // x part
    gemm_acc(hpHi, hpLo, whhHi, whhLo, rowA, wrow0, wrow1, acc0, acc1);   // h part

    // C layout: VGPR r -> M = mbase + r, N = lane%16 (lanes>=16: M += 8).
    const int mbase = m_tile * 16 + ((lane & 16) ? 8 : 0);
    const int n0 = g0 * 16 + (lane & 15);
    const int n1 = g1 * 16 + (lane & 15);
    float* d0 = &gate_lds[n0 * 68 + mbase];
    float* d1 = &gate_lds[n1 * 68 + mbase];
    *(float4*)(d0)     = make_float4(acc0[0], acc0[1], acc0[2], acc0[3]);
    *(float4*)(d0 + 4) = make_float4(acc0[4], acc0[5], acc0[6], acc0[7]);
    *(float4*)(d1)     = make_float4(acc1[0], acc1[1], acc1[2], acc1[3]);
    *(float4*)(d1 + 4) = make_float4(acc1[4], acc1[5], acc1[6], acc1[7]);
    __syncthreads();

    // Pointwise: thread handles 4 (batch, j_local) pairs; c stays in registers.
#pragma unroll
    for (int q = 0; q < 4; ++q) {
        int p = tid + 256 * q;
        int j = p & 15;
        int b = p >> 4;
        float ig = gate_lds[(0 * 16 + j) * 68 + b] + bias[0 * HID + j0 + j];
        float fg = gate_lds[(1 * 16 + j) * 68 + b] + bias[1 * HID + j0 + j];
        float gg = gate_lds[(2 * 16 + j) * 68 + b] + bias[2 * HID + j0 + j];
        float og = gate_lds[(3 * 16 + j) * 68 + b] + bias[3 * HID + j0 + j];
        float i_ = 1.f / (1.f + __expf(-ig));
        float f_ = 1.f / (1.f + __expf(-fg));
        float g_ = tanhf(gg);
        float o_ = 1.f / (1.f + __expf(-og));
        float c  = f_ * c_reg[q] + i_ * g_;
        c_reg[q] = c;
        float h  = o_ * tanhf(c);
        unsigned short hh, hl;
        splitbf(h, hh, hl);
        int idx = b * HID + j0 + j;
        houtHi[idx] = hh;
        houtLo[idx] = hl;
        if (foutOpt) foutOpt[idx] = h;
    }
    __syncthreads();
}

// ---------------------------------------------------------------------------
// Workspace layout (ushort elements after 256-byte header):
//   [0..4NE)       x staging: xsHi[0], xsLo[0], xsHi[1], xsLo[1]
//   [4NE..12NE)    h buffers: (layer, buf, hi/lo) -> 4NE + (l*4 + buf*2 + hl)*NE
//   [12NE..)       weights bf16: (layer*4 + which*2 + hl)*WE  which: 0=wih 1=whh
// ---------------------------------------------------------------------------

__global__ void k_convert_weights(const float* __restrict__ wih0,
                                  const float* __restrict__ whh0,
                                  const float* __restrict__ wih1,
                                  const float* __restrict__ whh1,
                                  unsigned short* __restrict__ base) {
    unsigned short* W = base + 12 * NE;
    const float* srcs[4] = {wih0, whh0, wih1, whh1};
    int g = blockIdx.x * blockDim.x + threadIdx.x;
    int stride = gridDim.x * blockDim.x;
    for (int e = g; e < WE; e += stride) {
#pragma unroll
        for (int m = 0; m < 4; ++m) {
            int layer = m >> 1, which = m & 1;
            unsigned short hi, lo;
            splitbf(srcs[m][e], hi, lo);
            W[(layer * 4 + which * 2 + 0) * WE + e] = hi;
            W[(layer * 4 + which * 2 + 1) * WE + e] = lo;
        }
    }
}

__global__ void k_init(const float* __restrict__ x, unsigned short* __restrict__ base,
                       unsigned* __restrict__ bar) {
    int g = blockIdx.x * blockDim.x + threadIdx.x;
    if (g < 2) bar[g] = 0;                        // barrier count + generation
    if (g < NE) {
        // h_prev for step 0 lives in buf 1: h = 1.0 (bf16 hi = 0x3F80, lo = 0)
        base[4 * NE + (0 * 4 + 1 * 2 + 0) * NE + g] = 0x3F80;
        base[4 * NE + (0 * 4 + 1 * 2 + 1) * NE + g] = 0;
        base[4 * NE + (1 * 4 + 1 * 2 + 0) * NE + g] = 0x3F80;
        base[4 * NE + (1 * 4 + 1 * 2 + 1) * NE + g] = 0;
        // stage x_0 into xs buf 0
        unsigned short hi, lo;
        splitbf(x[g], hi, lo);
        base[0 * NE + g] = hi;
        base[1 * NE + g] = lo;
    }
}

__global__ __launch_bounds__(256)
void k_lstm_persistent(const float* __restrict__ x,
                       const float* __restrict__ bias0,
                       const float* __restrict__ bias1,
                       float* __restrict__ out,
                       unsigned short* __restrict__ base,
                       unsigned* __restrict__ bar) {
    __shared__ float gate_lds[64 * 68];          // padded to kill bank conflicts

    const int layer = blockIdx.x >> 6;           // 0 or 1
    const int bIdx  = blockIdx.x & 63;           // owner of 16 hidden columns
    const int j0    = bIdx * 16;

    unsigned short* xsHi[2] = {base + 0 * NE, base + 2 * NE};
    unsigned short* xsLo[2] = {base + 1 * NE, base + 3 * NE};
    unsigned short* hB = base + 4 * NE;
#define H_HI(l, buf) (hB + ((l) * 4 + (buf) * 2 + 0) * NE)
#define H_LO(l, buf) (hB + ((l) * 4 + (buf) * 2 + 1) * NE)
    unsigned short* W = base + 12 * NE;
    unsigned short* wihHi = W + (layer * 4 + 0) * WE;
    unsigned short* wihLo = W + (layer * 4 + 1) * WE;
    unsigned short* whhHi = W + (layer * 4 + 2) * WE;
    unsigned short* whhLo = W + (layer * 4 + 3) * WE;
    const float* bias = (layer == 0) ? bias0 : bias1;

    float c_reg[4] = {0.f, 0.f, 0.f, 0.f};

    for (int i = 0; i <= T_STEPS; ++i) {
        if (layer == 0) {
            if (i < T_STEPS) {
                const int s = i;
                lstm_step(xsHi[s & 1], xsLo[s & 1],
                          H_HI(0, (s + 1) & 1), H_LO(0, (s + 1) & 1),
                          wihHi, wihLo, whhHi, whhLo, bias, j0,
                          H_HI(0, s & 1), H_LO(0, s & 1),
                          nullptr, gate_lds, c_reg);
                // Stage x_{s+1} as split-bf16 (64 blocks x 256 threads x 4 elems)
                if (s + 1 < T_STEPS) {
                    int e0 = (bIdx * 256 + threadIdx.x) * 4;
                    float4 v = *(const float4*)(x + (s + 1) * NE + e0);
                    unsigned short h0, l0, h1, l1, h2, l2, h3, l3;
                    splitbf(v.x, h0, l0); splitbf(v.y, h1, l1);
                    splitbf(v.z, h2, l2); splitbf(v.w, h3, l3);
                    ushort4 hv = {h0, h1, h2, h3};
                    ushort4 lv = {l0, l1, l2, l3};
                    *(ushort4*)(xsHi[(s + 1) & 1] + e0) = hv;
                    *(ushort4*)(xsLo[(s + 1) & 1] + e0) = lv;
                }
            }
        } else {
            if (i >= 1) {
                const int s = i - 1;                       // pipelined one step behind
                lstm_step(H_HI(0, s & 1), H_LO(0, s & 1),  // h_s from layer 0
                          H_HI(1, (s + 1) & 1), H_LO(1, (s + 1) & 1),
                          wihHi, wihLo, whhHi, whhLo, bias, j0,
                          H_HI(1, s & 1), H_LO(1, s & 1),
                          out + s * NE, gate_lds, c_reg);
            }
        }
        gridBarrier(bar, bar + 1, 128u);
    }
#undef H_HI
#undef H_LO
}

extern "C" void kernel_launch(void* const* d_in, const int* in_sizes, int n_in,
                              void* d_out, int out_size, void* d_ws, size_t ws_size,
                              hipStream_t stream) {
    const float* x    = (const float*)d_in[0];   // [T,B,IN]
    const float* wih0 = (const float*)d_in[1];   // [4H,IN]
    const float* whh0 = (const float*)d_in[2];   // [4H,H]
    const float* b0   = (const float*)d_in[3];   // [4H]
    const float* wih1 = (const float*)d_in[4];   // [4H,H]
    const float* whh1 = (const float*)d_in[5];   // [4H,H]
    const float* b1   = (const float*)d_in[6];   // [4H]
    float* out = (float*)d_out;                  // [T,B,H]

    unsigned* bar        = (unsigned*)d_ws;                       // 2 words
    unsigned short* base = (unsigned short*)((char*)d_ws + 256);  // data arrays

    (void)in_sizes; (void)n_in; (void)out_size; (void)ws_size;

    k_convert_weights<<<4096, 256, 0, stream>>>(wih0, whh0, wih1, whh1, base);
    k_init<<<256, 256, 0, stream>>>(x, base, bar);
    k_lstm_persistent<<<128, 256, 0, stream>>>(x, b0, b1, out, base, bar);
}
